// Block_16
// MI455X (gfx1250) — compile-verified, matrix operand order checked by reading, not run on hardware
//
#include <hip/hip_runtime.h>
#include <hip/hip_bf16.h>

// ---------------------------------------------------------------------------
// Transformer block for MI455X (gfx1250): bf16 WMMA + async global->LDS.
// B=512 T=128 C=384 H=8 D=48 FF=1536
// ---------------------------------------------------------------------------

#define BB 512
#define TT 128
#define CC 384
#define HH 8
#define DD 48
#define FFD 1536
#define MM (BB * TT)          // 65536 token rows
#define QKVN (3 * CC)         // 1152

#define LDS_AS __attribute__((address_space(3)))
#define GLB_AS __attribute__((address_space(1)))

typedef __attribute__((ext_vector_type(16))) __bf16 v16bf;
typedef __attribute__((ext_vector_type(8)))  __bf16 v8bf;
typedef __attribute__((ext_vector_type(8)))  float  v8f;
typedef int v4i __attribute__((vector_size(16)));

#if __has_builtin(__builtin_amdgcn_global_load_async_to_lds_b128) && \
    __has_builtin(__builtin_amdgcn_s_wait_asynccnt)
#define HAVE_ASYNC_LDS 1
#else
#define HAVE_ASYNC_LDS 0
#endif

// 16-byte global -> LDS copy. Async (ASYNCcnt-tracked) when available.
__device__ inline void cp_b128(LDS_AS __bf16* lds, const __bf16* g) {
#if HAVE_ASYNC_LDS
  __builtin_amdgcn_global_load_async_to_lds_b128(
      (GLB_AS v4i*)g, (LDS_AS v4i*)lds, 0, 0);
#else
  *(LDS_AS v8bf*)lds = *(const v8bf*)g;
#endif
}

__device__ inline void wait_cp4(int pending) {
#if HAVE_ASYNC_LDS
  if (pending) __builtin_amdgcn_s_wait_asynccnt(4);   // double-buffered slabs
  else         __builtin_amdgcn_s_wait_asynccnt(0);
#else
  (void)pending;
#endif
}
__device__ inline void wait_cp1(int pending) {
#if HAVE_ASYNC_LDS
  if (pending) __builtin_amdgcn_s_wait_asynccnt(1);
  else         __builtin_amdgcn_s_wait_asynccnt(0);
#else
  (void)pending;
#endif
}

// ---------------------------------------------------------------------------
// WMMA helpers (wave32). 16x16x32 bf16 -> f32 accumulate.
// A fragment (16x32 row-major, "row" = M): lane 0-15 row=lane, kh=0;
// lane 16-31 row=lane-16, kh=8; elements kh+0..7 and kh+16..23.
// B fragment loaded from a TRANSPOSED matrix (N x K row-major), same pattern.
// ---------------------------------------------------------------------------

// Fragment from GLOBAL memory (generic ptr -> inferred global_load_b128)
__device__ inline v16bf load_frag(const __bf16* __restrict__ base, int row0,
                                  int ld, int k0) {
  int lane = threadIdx.x & 31;
  int r    = lane & 15;
  int kh   = (lane >> 4) << 3;   // 0 or 8
  const __bf16* p = base + (long)(row0 + r) * ld + k0 + kh;
  v8bf lo = *(const v8bf*)(p);
  v8bf hi = *(const v8bf*)(p + 16);
  v16bf out;
#pragma unroll
  for (int j = 0; j < 8; ++j) { out[j] = lo[j]; out[8 + j] = hi[j]; }
  return out;
}

// Fragment from LDS (addrspace(3)-typed -> ds_load_b128)
__device__ inline v16bf load_frag_l(const LDS_AS __bf16* base, int row0,
                                    int ld, int k0) {
  int lane = threadIdx.x & 31;
  int r    = lane & 15;
  int kh   = (lane >> 4) << 3;   // 0 or 8
  const LDS_AS __bf16* p = base + (row0 + r) * ld + k0 + kh;
  v8bf lo = *(const LDS_AS v8bf*)(p);
  v8bf hi = *(const LDS_AS v8bf*)(p + 16);
  v16bf out;
#pragma unroll
  for (int j = 0; j < 8; ++j) { out[j] = lo[j]; out[8 + j] = hi[j]; }
  return out;
}

__device__ inline v8f wmma_bf16(v16bf a, v16bf b, v8f c) {
  return __builtin_amdgcn_wmma_f32_16x16x32_bf16(false, a, false, b,
                                                 (short)0, c, false, false);
}

// C/D layout: lane 0-15 -> N = lane, M = r ; lane 16-31 -> N = lane-16, M = 8+r

// ---------------------------------------------------------------------------
// Elementwise prep kernels
// ---------------------------------------------------------------------------
__global__ __launch_bounds__(256) void f32_to_bf16_k(const float* __restrict__ src,
                                                     __bf16* __restrict__ dst, int n) {
  int i = blockIdx.x * 256 + threadIdx.x;
  if (i < n) dst[i] = (__bf16)src[i];
}

// dst [N][K] bf16 = transpose of src [K][N] f32
__global__ __launch_bounds__(256) void transpose_to_bf16_k(const float* __restrict__ src,
                                                           __bf16* __restrict__ dst,
                                                           int K, int N) {
  int i = blockIdx.x * 256 + threadIdx.x;
  if (i < K * N) {
    int n = i / K, k = i % K;
    dst[i] = (__bf16)src[(long)k * N + n];
  }
}

// Pack Wq|Wk|Wv (each [C][C]) into one transposed [1152][384] bf16
__global__ __launch_bounds__(256) void pack_wqkvT_k(const float* __restrict__ Wq,
                                                    const float* __restrict__ Wk,
                                                    const float* __restrict__ Wv,
                                                    __bf16* __restrict__ dst) {
  int i = blockIdx.x * 256 + threadIdx.x;
  if (i < QKVN * CC) {
    int n = i / CC, k = i % CC;
    const float* W = (n < CC) ? Wq : (n < 2 * CC) ? Wk : Wv;
    dst[i] = (__bf16)W[(long)k * CC + (n % CC)];
  }
}

// ---------------------------------------------------------------------------
// bf16 GEMM: C[M x N] = A[M x K] * BT[N x K]^T (+bias, optional relu).
// 256 threads (8 waves), 128x128 output tile per workgroup.
// Double-buffered 128x32 A/B slabs staged in LDS via async copies; each wave
// register-blocks 2x4 (32x64) accumulators -> 8 WMMAs per K-slab.
// ---------------------------------------------------------------------------
__global__ __launch_bounds__(256) void gemm_bf16_k(const __bf16* __restrict__ A,
                                                   const __bf16* __restrict__ BT,
                                                   const float* __restrict__ bias,
                                                   __bf16* __restrict__ C,
                                                   int K, int ldc, int relu) {
  extern __shared__ char smem[];
  LDS_AS __bf16* Ab0 = (LDS_AS __bf16*)smem;     // 2 x [128][32]
  LDS_AS __bf16* Bb0 = Ab0 + 2 * 128 * 32;       // 2 x [128][32]

  int m0  = blockIdx.x * 128;
  int n0  = blockIdx.y * 128;
  int tid = threadIdx.x;
  int w = tid >> 5, lane = tid & 31;
  int r0 = tid >> 2;            // 0..63
  int c0 = (tid & 3) * 8;       // 0/8/16/24

  auto stage = [&](int p, int k0) {
    LDS_AS __bf16* Abp = Ab0 + p * (128 * 32);
    LDS_AS __bf16* Bbp = Bb0 + p * (128 * 32);
    cp_b128(Abp + r0 * 32 + c0,        A  + (long)(m0 + r0)      * K + k0 + c0);
    cp_b128(Abp + (64 + r0) * 32 + c0, A  + (long)(m0 + 64 + r0) * K + k0 + c0);
    cp_b128(Bbp + r0 * 32 + c0,        BT + (long)(n0 + r0)      * K + k0 + c0);
    cp_b128(Bbp + (64 + r0) * 32 + c0, BT + (long)(n0 + 64 + r0) * K + k0 + c0);
  };

  int nslabs = K >> 5;
  stage(0, 0);

  v8f acc[2][4] = {};
  int mtb = (w & 3) * 32;       // wave row offset within tile
  int ntb = (w >> 2) * 64;      // wave col offset within tile

  for (int s = 0; s < nslabs; ++s) {
    int p = s & 1;
    int more = (s + 1 < nslabs);
    if (more) stage(p ^ 1, (s + 1) << 5);
    wait_cp4(more);
    __syncthreads();
    LDS_AS __bf16* Abp = Ab0 + p * (128 * 32);
    LDS_AS __bf16* Bbp = Bb0 + p * (128 * 32);
    v16bf af0 = load_frag_l(Abp, mtb,      32, 0);
    v16bf af1 = load_frag_l(Abp, mtb + 16, 32, 0);
#pragma unroll
    for (int j = 0; j < 4; ++j) {
      v16bf bf = load_frag_l(Bbp, ntb + j * 16, 32, 0);
      acc[0][j] = wmma_bf16(af0, bf, acc[0][j]);
      acc[1][j] = wmma_bf16(af1, bf, acc[1][j]);
    }
    __syncthreads();
  }

  int n  = lane & 15;
  int mh = (lane >> 4) << 3;
#pragma unroll
  for (int i = 0; i < 2; ++i) {
#pragma unroll
    for (int j = 0; j < 4; ++j) {
      int nbase = n0 + ntb + j * 16 + n;
      float bv = bias ? bias[nbase] : 0.0f;
#pragma unroll
      for (int r = 0; r < 8; ++r) {
        float xv = acc[i][j][r] + bv;
        if (relu) xv = fmaxf(xv, 0.0f);
        C[(long)(m0 + mtb + i * 16 + mh + r) * ldc + nbase] = (__bf16)xv;
      }
    }
  }
}

// ---------------------------------------------------------------------------
// Causal attention, one workgroup per (b,h). LDS-resident Q,K,V^T,S,P.
// ---------------------------------------------------------------------------
__global__ __launch_bounds__(256) void attention_k(const __bf16* __restrict__ qkv,
                                                   __bf16* __restrict__ attnb) {
  extern __shared__ char smem[];
  LDS_AS __bf16* qs  = (LDS_AS __bf16*)smem;       // [128][64] (pad 48->64, 0s)
  LDS_AS __bf16* ks  = qs  + 128 * 64;             // [128][64]
  LDS_AS __bf16* vsT = ks  + 128 * 64;             // [48][128] (V transposed)
  LDS_AS float*  ss  = (LDS_AS float*)(vsT + 48 * 128);  // [128][128] f32
  LDS_AS __bf16* ps  = (LDS_AS __bf16*)(ss + 128 * 128); // [128][128] bf16

  int b = blockIdx.x >> 3;
  int h = blockIdx.x & 7;
  const __bf16* base = qkv + (long)b * TT * QKVN + h * DD;

  for (int idx = threadIdx.x; idx < 128 * 64; idx += 256) {
    int t = idx >> 6, d = idx & 63;
    __bf16 z = (__bf16)0.0f;
    qs[idx] = (d < DD) ? base[(long)t * QKVN + d]      : z;
    ks[idx] = (d < DD) ? base[(long)t * QKVN + CC + d] : z;
  }
  for (int idx = threadIdx.x; idx < DD * 128; idx += 256) {
    int d = idx >> 7, t = idx & 127;
    vsT[idx] = base[(long)t * QKVN + 2 * CC + d];
  }
  __syncthreads();

  int w    = threadIdx.x >> 5;
  int lane = threadIdx.x & 31;
  const float scale = 0.14433756729740643f;  // 1/sqrt(48)

  // S = Q @ K^T (wave w owns row-tile w; causal tiles nt <= w only)
  for (int nt = 0; nt <= w; ++nt) {
    v8f acc = {};
#pragma unroll
    for (int k0 = 0; k0 < 64; k0 += 32) {
      v16bf a = load_frag_l(qs, w * 16,  64, k0);
      v16bf b = load_frag_l(ks, nt * 16, 64, k0);
      acc = wmma_bf16(a, b, acc);
    }
    int n  = lane & 15;
    int mh = (lane >> 4) << 3;
#pragma unroll
    for (int r = 0; r < 8; ++r)
      ss[(w * 16 + mh + r) * 128 + nt * 16 + n] = acc[r] * scale;
  }
  __syncthreads();

  if (threadIdx.x < 128) {
    int t = threadIdx.x;
    float mx = -1e30f;
    for (int s = 0; s <= t; ++s) mx = fmaxf(mx, ss[t * 128 + s]);
    float sum = 0.0f;
    for (int s = 0; s <= t; ++s) {
      float e = __expf(ss[t * 128 + s] - mx);
      ss[t * 128 + s] = e;
      sum += e;
    }
    float inv = 1.0f / sum;
    for (int s = 0; s <= t; ++s)  ps[t * 128 + s] = (__bf16)(ss[t * 128 + s] * inv);
    for (int s = t + 1; s < 128; ++s) ps[t * 128 + s] = (__bf16)0.0f;
  }
  __syncthreads();

  // O = P @ V
#pragma unroll
  for (int nt = 0; nt < 3; ++nt) {
    v8f acc = {};
#pragma unroll
    for (int k0 = 0; k0 < 128; k0 += 32) {
      v16bf a = load_frag_l(ps,  w * 16,  128, k0);
      v16bf b = load_frag_l(vsT, nt * 16, 128, k0);
      acc = wmma_bf16(a, b, acc);
    }
    int n  = lane & 15;
    int mh = (lane >> 4) << 3;
#pragma unroll
    for (int r = 0; r < 8; ++r) {
      int t = w * 16 + mh + r;
      attnb[(long)(b * TT + t) * CC + h * DD + nt * 16 + n] = (__bf16)acc[r];
    }
  }
}

// ---------------------------------------------------------------------------
// GEMM (N fixed = 384) + bias + residual + LayerNorm, fused.
// A (activations) staged through double-buffered async LDS slabs, 12 live
// accumulators per wave (A fragment reused 12x); weights stream from L2.
// ---------------------------------------------------------------------------
__global__ __launch_bounds__(256) void gemm_residual_ln_k(
    const __bf16* __restrict__ A, const __bf16* __restrict__ BT,
    const float* __restrict__ bias, const float* __restrict__ resid,
    const float* __restrict__ g, const float* __restrict__ beta,
    float* __restrict__ outF, __bf16* __restrict__ outB, int K) {
  extern __shared__ char smem[];
  LDS_AS float*  cbuf = (LDS_AS float*)smem;                      // [64][384]
  LDS_AS __bf16* Ab0  = (LDS_AS __bf16*)(cbuf + 64 * CC);         // 2 x [64][32]

  int m0  = blockIdx.x * 64;
  int tid = threadIdx.x;
  int w = tid >> 5, lane = tid & 31;
  int r0 = tid >> 2;            // 0..63
  int c0 = (tid & 3) * 8;

  auto stageA = [&](int p, int k0) {
    cp_b128(Ab0 + p * (64 * 32) + r0 * 32 + c0,
            A + (long)(m0 + r0) * K + k0 + c0);
  };

  int nslabs = K >> 5;
  stageA(0, 0);

  v8f acc[12] = {};
  int mtb = (w & 3) * 16;

  for (int s = 0; s < nslabs; ++s) {
    int p = s & 1;
    int more = (s + 1 < nslabs);
    if (more) stageA(p ^ 1, (s + 1) << 5);
    wait_cp1(more);
    __syncthreads();
    v16bf af = load_frag_l(Ab0 + p * (64 * 32), mtb, 32, 0);
    int k0 = s << 5;
#pragma unroll
    for (int i = 0; i < 12; ++i) {
      int nt = (w >> 2) + 2 * i;                      // 0..23
      v16bf bf = load_frag(BT, nt * 16, K, k0);       // weights from global/L2
      acc[i] = wmma_bf16(af, bf, acc[i]);
    }
    __syncthreads();
  }

  int n  = lane & 15;
  int mh = (lane >> 4) << 3;
#pragma unroll
  for (int i = 0; i < 12; ++i) {
    int nb = ((w >> 2) + 2 * i) * 16;
    float bv = bias[nb + n];
#pragma unroll
    for (int r = 0; r < 8; ++r)
      cbuf[(mtb + mh + r) * CC + nb + n] = acc[i][r] + bv;
  }
  __syncthreads();

  // residual + LayerNorm, one row (384 cols) per thread, threads 0..63
  if (threadIdx.x < 64) {
    int row = threadIdx.x;
    long gr = (long)(m0 + row) * CC;
    float mu = 0.0f;
    for (int c = 0; c < CC; ++c) {
      float v = cbuf[row * CC + c] + resid[gr + c];
      cbuf[row * CC + c] = v;
      mu += v;
    }
    mu *= (1.0f / CC);
    float var = 0.0f;
    for (int c = 0; c < CC; ++c) {
      float d = cbuf[row * CC + c] - mu;
      var += d * d;
    }
    var *= (1.0f / CC);
    float rs = rsqrtf(var + 1e-5f);
    for (int c = 0; c < CC; ++c) {
      float o = (cbuf[row * CC + c] - mu) * rs * g[c] + beta[c];
      outF[gr + c] = o;
      if (outB) outB[gr + c] = (__bf16)o;
    }
  }
}

// ---------------------------------------------------------------------------
// Host launcher
// ---------------------------------------------------------------------------
extern "C" void kernel_launch(void* const* d_in, const int* in_sizes, int n_in,
                              void* d_out, int out_size, void* d_ws, size_t ws_size,
                              hipStream_t stream) {
  const float* x     = (const float*)d_in[0];
  const float* Wq    = (const float*)d_in[1];
  const float* Wk    = (const float*)d_in[2];
  const float* Wv    = (const float*)d_in[3];
  const float* Wo    = (const float*)d_in[4];
  const float* bo    = (const float*)d_in[5];
  const float* W1    = (const float*)d_in[6];
  const float* b1    = (const float*)d_in[7];
  const float* W2    = (const float*)d_in[8];
  const float* b2    = (const float*)d_in[9];
  const float* g1    = (const float*)d_in[10];
  const float* beta1 = (const float*)d_in[11];
  const float* g2    = (const float*)d_in[12];
  const float* beta2 = (const float*)d_in[13];
  float* out = (float*)d_out;

  char* ws = (char*)d_ws;
  size_t off = 0;
  auto alloc = [&](size_t bytes) -> void* {
    void* p = ws + off;
    off += (bytes + 255) & ~(size_t)255;
    return p;
  };

  __bf16* xb     = (__bf16*)alloc((size_t)MM * CC * 2);
  __bf16* qkv    = (__bf16*)alloc((size_t)MM * QKVN * 2);
  __bf16* attnb  = (__bf16*)alloc((size_t)MM * CC * 2);
  float*  x1f    = (float*) alloc((size_t)MM * CC * 4);
  __bf16* x1b    = (__bf16*)alloc((size_t)MM * CC * 2);
  __bf16* hbuf   = (__bf16*)alloc((size_t)MM * FFD * 2);
  __bf16* WqkvT  = (__bf16*)alloc((size_t)QKVN * CC * 2);
  __bf16* WoT    = (__bf16*)alloc((size_t)CC * CC * 2);
  __bf16* W1T    = (__bf16*)alloc((size_t)FFD * CC * 2);
  __bf16* W2T    = (__bf16*)alloc((size_t)CC * FFD * 2);

  // ---- prep: bf16 conversion + weight transposes ----
  {
    int n = MM * CC;
    f32_to_bf16_k<<<(n + 255) / 256, 256, 0, stream>>>(x, xb, n);
  }
  pack_wqkvT_k<<<(QKVN * CC + 255) / 256, 256, 0, stream>>>(Wq, Wk, Wv, WqkvT);
  transpose_to_bf16_k<<<(CC * CC + 255) / 256, 256, 0, stream>>>(Wo, WoT, CC, CC);
  transpose_to_bf16_k<<<(CC * FFD + 255) / 256, 256, 0, stream>>>(W1, W1T, CC, FFD);
  transpose_to_bf16_k<<<(FFD * CC + 255) / 256, 256, 0, stream>>>(W2, W2T, FFD, CC);

  size_t gemm_smem = (size_t)4 * 128 * 32 * 2;               // 32 KB
  size_t ln_smem   = (size_t)64 * CC * 4 + 2 * 64 * 32 * 2;  // 104 KB

  // ---- QKV projection: [M,384] @ [384,1152] -> qkv bf16 ----
  gemm_bf16_k<<<dim3(MM / 128, QKVN / 128), 256, gemm_smem, stream>>>(
      xb, WqkvT, nullptr, qkv, CC, QKVN, 0);

  // ---- causal attention per (b,h) ----
  {
    size_t smem = (size_t)(128 * 64 + 128 * 64 + 48 * 128) * 2
                + (size_t)128 * 128 * 4
                + (size_t)128 * 128 * 2;
    attention_k<<<BB * HH, 256, smem, stream>>>(qkv, attnb);
  }

  // ---- output projection + residual + LN1 ----
  gemm_residual_ln_k<<<MM / 64, 256, ln_smem, stream>>>(
      attnb, WoT, bo, x, g1, beta1, x1f, x1b, CC);

  // ---- FFN up + relu: [M,384] @ [384,1536] ----
  gemm_bf16_k<<<dim3(MM / 128, FFD / 128), 256, gemm_smem, stream>>>(
      x1b, W1T, b1, hbuf, CC, FFD, 1);

  // ---- FFN down + residual + LN2 -> d_out ----
  gemm_residual_ln_k<<<MM / 64, 256, ln_smem, stream>>>(
      hbuf, W2T, b2, x1f, g2, beta2, out, nullptr, FFD);
}
